// BiAttention_26147760898610
// MI455X (gfx1250) — matrix operand order repaired and checked by reading, not compiled, not run on hardware
//
#include <hip/hip_runtime.h>
#include <hip/hip_bf16.h>
#include <stdint.h>

// ---------------------------------------------------------------------------
// BiAttention (mode==1) fused kernels for gfx1250 (MI455X), f16 WMMA path.
//   input1: [16,2048,300] f32   input2: [16,1024,300] f32
//   W2/W3: [300,300] f32, b2/b3: [300] f32
//   out: 2 x [16,2048,300] f32 concatenated
// All operands are zero-padded to D=320 (multiple of WMMA K=32) in f16
// workspace so every GEMM inner loop is branchless b128 loads + v_wmma.
// ---------------------------------------------------------------------------

typedef __attribute__((ext_vector_type(16))) _Float16 v16h;
typedef __attribute__((ext_vector_type(8)))  float    v8f;
typedef __attribute__((ext_vector_type(4)))  int      v4i;

#define Dk    300
#define DIMP  320     // D padded to multiple of 32 (WMMA K), pad region zeroed
#define NB    16
#define T1    2048
#define T2    1024

union HV16 { uint4 u[2]; v16h v; };
union HU8  { uint4 u;    _Float16 h[8]; };

#define WMMA_F16(a, b, c) \
  __builtin_amdgcn_wmma_f32_16x16x32_f16(false, (a), false, (b), (short)0, (c), false, false)

// gfx1250 async global->LDS copies (ASYNCcnt-tracked, no VGPR round-trip).
// Builtin signature (from probe): (v4i addrspace(1)*, v4i addrspace(3)*, imm, imm).
#if defined(__gfx1250__) &&                                            \
    __has_builtin(__builtin_amdgcn_global_load_async_to_lds_b128) &&   \
    __has_builtin(__builtin_amdgcn_s_wait_asynccnt)
#define HAVE_ASYNC_LDS 1
#define AS_GLOBAL_V4(p) ((__attribute__((address_space(1))) v4i*)(p))
#define AS_LDS_V4(p)    ((__attribute__((address_space(3))) v4i*)(p))
#else
#define HAVE_ASYNC_LDS 0
#endif

// ds_swizzle group-of-32: offset = xor<<10 | or<<5 | and ; and=0x1f keeps full group
#define SWZ(v, imm) __int_as_float(__builtin_amdgcn_ds_swizzle(__float_as_int(v), (imm)))

__device__ __forceinline__ float redmax16(float v) {
  // reduce across the 16 lanes of this lane's half (xor masks < 16 stay in-half)
  v = fmaxf(v, SWZ(v, 0x041F));
  v = fmaxf(v, SWZ(v, 0x081F));
  v = fmaxf(v, SWZ(v, 0x101F));
  v = fmaxf(v, SWZ(v, 0x201F));
  return v;
}
__device__ __forceinline__ float redsum16(float v) {
  v += SWZ(v, 0x041F);
  v += SWZ(v, 0x081F);
  v += SWZ(v, 0x101F);
  v += SWZ(v, 0x201F);
  return v;
}

// ---------------------------------------------------------------------------
// f32 [rows,300] -> f16 [rows,320], zero pad (so WMMA K=320 is exact)
// ---------------------------------------------------------------------------
__global__ void cvt_pad_kernel(const float* __restrict__ in,
                               _Float16* __restrict__ out, int rows) {
  int i = blockIdx.x * blockDim.x + threadIdx.x;
  int total = rows * DIMP;
  if (i >= total) return;
  int r = i / DIMP, d = i - r * DIMP;
  float v = (d < Dk) ? in[(size_t)r * Dk + d] : 0.0f;
  out[i] = (_Float16)v;
}

// W [300][300] f32 -> Wp [320][320] f16, zero-padded both dims.
__global__ void cvt_w_kernel(const float* __restrict__ W,
                             _Float16* __restrict__ Wp) {
  int i = blockIdx.x * blockDim.x + threadIdx.x;
  if (i >= DIMP * DIMP) return;
  int e = i / DIMP, d = i - e * DIMP;
  float v = (e < Dk && d < Dk) ? W[(size_t)e * Dk + d] : 0.0f;
  Wp[i] = (_Float16)v;
}

// bias [300] f32 -> bp [320] f32, zero-padded.
__global__ void cvt_bias_kernel(const float* __restrict__ b,
                                float* __restrict__ bp) {
  int i = threadIdx.x;
  if (i < DIMP) bp[i] = (i < Dk) ? b[i] : 0.0f;
}

// ---------------------------------------------------------------------------
// proj[r,e] = bias[e] + sum_d attn[r,d] * W[e,d]   (r = b*1024+s), f16 out.
// One wave per 16(s) x 16(e) tile; A from f16 attn_h, B from padded f16 Wp.
// Padded Wp rows e>=300 are zero => proj pad region is exactly zero.
// ---------------------------------------------------------------------------
__global__ void __launch_bounds__(128)
proj_kernel(const _Float16* __restrict__ attn_h,  // [NB*T2][DIMP]
            const _Float16* __restrict__ Wp,      // [DIMP][DIMP] padded f16
            const float* __restrict__ bp,         // [DIMP] padded bias
            _Float16* __restrict__ proj_h)        // [NB*T2][DIMP]
{
  const int lane = threadIdx.x & 31;
  const int wave = threadIdx.x >> 5;
  const int et   = blockIdx.y * 4 + wave;   // 0..19 e-tiles (cover 320)
  const int r0   = blockIdx.x * 16;         // s-row tile (never straddles b)
  const int col  = lane & 15;
  const int hl   = lane >> 4;
  const int e    = et * 16 + col;

  const _Float16* aptr = attn_h + (size_t)(r0 + col) * DIMP;
  const _Float16* wrow = Wp + (size_t)e * DIMP;   // B: b[i] = Wp[e][k]

  v8f c = {};
  #pragma unroll
  for (int kk = 0; kk < 10; ++kk) {
    const int kb = kk * 32;
    HV16 a;                                  // A layout: row=lane&15, K split by half
    a.u[0] = *(const uint4*)(aptr + kb + hl * 16);
    a.u[1] = *(const uint4*)(aptr + kb + hl * 16 + 8);
    HV16 b;                                  // B layout: n=lane&15, K=kb+hl*16+i
    b.u[0] = *(const uint4*)(wrow + kb + hl * 16);
    b.u[1] = *(const uint4*)(wrow + kb + hl * 16 + 8);
    c = WMMA_F16(a.v, b.v, c);
  }
  const float bv = bp[e];                    // zero in pad region
  #pragma unroll
  for (int j = 0; j < 8; ++j) {              // C layout: row=j+8*hl, col=lane&15
    proj_h[(size_t)(r0 + j + 8 * hl) * DIMP + et * 16 + col] =
        (_Float16)(c[j] + bv);
  }
}

// ---------------------------------------------------------------------------
// Fused flash attention. Block = 128 thr (4 waves), each wave owns 16 T-rows.
// Streams S in chunks of 32: scores (WMMA) -> online softmax (ds_swizzle
// reductions) -> P through LDS into A-layout -> 20 WMMAs into 16x320 fp32
// accumulator held in VGPRs. grid = (T1/64, B, 2{W2,W3}).
// ---------------------------------------------------------------------------
__global__ void __launch_bounds__(128)
flash_attn_kernel(const _Float16* __restrict__ main_h,   // [NB*T1][DIMP]
                  const _Float16* __restrict__ attn_h,   // [NB*T2][DIMP]
                  const _Float16* __restrict__ proj2_h,  // [NB*T2][DIMP]
                  const _Float16* __restrict__ proj3_h,  // [NB*T2][DIMP]
                  float* __restrict__ out)               // [2][NB][T1][300]
{
  // padded LDS strides chosen for bank spread + 16B alignment
  __shared__ _Float16 projC[32 * 328];      // proj chunk, row-major [s][e]
  __shared__ _Float16 attnT[320 * 40];      // attn chunk transposed [d][s]
  __shared__ _Float16 Pbuf[4][16 * 40];     // per-wave probability staging

  const int b    = blockIdx.y;
  const int z    = blockIdx.z;
  const _Float16* proj_h = z ? proj3_h : proj2_h;
  float* outz = out + (size_t)z * NB * T1 * Dk;

  const int tid  = threadIdx.x;
  const int lane = tid & 31;
  const int wave = tid >> 5;
  const int hl   = lane >> 4;
  const int col  = lane & 15;

  const int t0 = blockIdx.x * 64 + wave * 16;     // this wave's T-tile base
  const _Float16* mrow  = main_h + ((size_t)b * T1 + t0 + col) * DIMP;
  const _Float16* projB = proj_h + (size_t)b * T2 * DIMP;
  const _Float16* attnB = attn_h + (size_t)b * T2 * DIMP;

  v8f O[20];
  v8f vz = {};
  #pragma unroll
  for (int dt = 0; dt < 20; ++dt) O[dt] = vz;
  float m[8], l[8];
  #pragma unroll
  for (int j = 0; j < 8; ++j) { m[j] = -1e30f; l[j] = 0.0f; }

  for (int sc = 0; sc < T2 / 32; ++sc) {
    const int s0 = sc * 32;
    __syncthreads();
    // ---- cooperative stage: proj chunk (row-major) + attn chunk (transposed)
    if (sc + 1 < T2 / 32) {                 // speculative prefetch of next chunk
      __builtin_prefetch(projB + (size_t)(s0 + 32) * DIMP + (size_t)tid * 80, 0, 1);
      __builtin_prefetch(attnB + (size_t)(s0 + 32) * DIMP + (size_t)tid * 80, 0, 1);
    }
#if HAVE_ASYNC_LDS
    // proj chunk: pure async DMA into LDS, tracked by ASYNCcnt.
    #pragma unroll
    for (int it = 0; it < 10; ++it) {       // 32 rows * 40 x16B / 128 threads
      int idx = it * 128 + tid;
      int r = idx / 40, c = idx % 40;
      __builtin_amdgcn_global_load_async_to_lds_b128(
          AS_GLOBAL_V4(projB + (size_t)(s0 + r) * DIMP + c * 8),
          AS_LDS_V4(&projC[r * 328 + c * 8]), 0, 0);
    }
#else
    #pragma unroll
    for (int it = 0; it < 10; ++it) {       // 32 rows * 40 uint4 / 128 threads
      int idx = it * 128 + tid;
      int r = idx / 40, c = idx % 40;
      uint4 v = *(const uint4*)(projB + (size_t)(s0 + r) * DIMP + c * 8);
      *(uint4*)(&projC[r * 328 + c * 8]) = v;
    }
#endif
    // attn chunk: element transpose, must round-trip through VGPRs.
    #pragma unroll
    for (int it = 0; it < 10; ++it) {
      int idx = it * 128 + tid;
      int r = idx / 40, c = idx % 40;
      HU8 v;
      v.u = *(const uint4*)(attnB + (size_t)(s0 + r) * DIMP + c * 8);
      #pragma unroll
      for (int i = 0; i < 8; ++i) attnT[(c * 8 + i) * 40 + r] = v.h[i];
    }
#if HAVE_ASYNC_LDS
    __builtin_amdgcn_s_wait_asynccnt(0);    // own async ops done before barrier
#endif
    __syncthreads();

    // ---- scores: two 16x16 tiles (s cols 0-15 and 16-31), K = 320
    v8f c0 = vz, c1 = vz;
    #pragma unroll
    for (int kk = 0; kk < 10; ++kk) {
      const int kb = kk * 32;
      HV16 a;
      a.u[0] = *(const uint4*)(mrow + kb + hl * 16);
      a.u[1] = *(const uint4*)(mrow + kb + hl * 16 + 8);
      HV16 b0, b1;                           // b[i] = proj[s=col][e=kb+hl*16+i]
      b0.u[0] = *(const uint4*)(&projC[col * 328 + kb + hl * 16]);
      b0.u[1] = *(const uint4*)(&projC[col * 328 + kb + hl * 16 + 8]);
      b1.u[0] = *(const uint4*)(&projC[(16 + col) * 328 + kb + hl * 16]);
      b1.u[1] = *(const uint4*)(&projC[(16 + col) * 328 + kb + hl * 16 + 8]);
      c0 = WMMA_F16(a.v, b0.v, c0);
      c1 = WMMA_F16(a.v, b1.v, c1);
    }

    // ---- online softmax (row stats per (vgpr j, lane-half); lanes hold cols)
    float alpha[8];
    #pragma unroll
    for (int j = 0; j < 8; ++j) {
      float mc = redmax16(fmaxf(c0[j], c1[j]));
      float mn = fmaxf(m[j], mc);
      alpha[j] = __expf(m[j] - mn);
      m[j] = mn;
      float p0 = __expf(c0[j] - mn);
      float p1 = __expf(c1[j] - mn);
      c0[j] = p0; c1[j] = p1;
      l[j] = l[j] * alpha[j] + redsum16(p0 + p1);
    }
    #pragma unroll
    for (int dt = 0; dt < 20; ++dt)
      #pragma unroll
      for (int j = 0; j < 8; ++j) O[dt][j] *= alpha[j];

    // ---- re-layout P (C/D layout -> A layout) via per-wave LDS staging
    _Float16* P = Pbuf[wave];
    #pragma unroll
    for (int j = 0; j < 8; ++j) {
      P[(j + 8 * hl) * 40 + col]      = (_Float16)c0[j];
      P[(j + 8 * hl) * 40 + 16 + col] = (_Float16)c1[j];
    }
    HV16 pa;                                 // A: row=lane&15, K split by half
    pa.u[0] = *(const uint4*)(&P[col * 40 + hl * 16]);
    pa.u[1] = *(const uint4*)(&P[col * 40 + hl * 16 + 8]);

    // ---- O += P (16x32) x attn (32x16-per-d-tile), 20 d-tiles
    #pragma unroll
    for (int dt = 0; dt < 20; ++dt) {
      HV16 bb;                               // b[i] = attnT[d=dt*16+col][hl*16+i]
      bb.u[0] = *(const uint4*)(&attnT[(dt * 16 + col) * 40 + hl * 16]);
      bb.u[1] = *(const uint4*)(&attnT[(dt * 16 + col) * 40 + hl * 16 + 8]);
      O[dt] = WMMA_F16(pa.v, bb.v, O[dt]);
    }
  }

  // ---- normalize and store (skip the d>=300 pad)
  #pragma unroll
  for (int j = 0; j < 8; ++j) l[j] = 1.0f / l[j];
  #pragma unroll
  for (int dt = 0; dt < 19; ++dt) {
    const int d = dt * 16 + col;
    if (d < Dk) {
      #pragma unroll
      for (int j = 0; j < 8; ++j) {
        const int t = t0 + j + 8 * hl;
        outz[((size_t)b * T1 + t) * Dk + d] = O[dt][j] * l[j];
      }
    }
  }
}

// ---------------------------------------------------------------------------
extern "C" void kernel_launch(void* const* d_in, const int* in_sizes, int n_in,
                              void* d_out, int out_size, void* d_ws, size_t ws_size,
                              hipStream_t stream) {
  const float* input1 = (const float*)d_in[0];  // [16,2048,300]
  const float* input2 = (const float*)d_in[1];  // [16,1024,300]
  const float* W2     = (const float*)d_in[2];
  const float* b2     = (const float*)d_in[3];
  const float* W3     = (const float*)d_in[4];
  const float* b3     = (const float*)d_in[5];
  float* out = (float*)d_out;

  _Float16* ws = (_Float16*)d_ws;
  const size_t nAttn = (size_t)NB * T2 * DIMP;   // 5.24M halfs
  const size_t nMain = (size_t)NB * T1 * DIMP;   // 10.5M halfs
  const size_t nW    = (size_t)DIMP * DIMP;      // 102.4K halfs
  _Float16* attn_h  = ws;
  _Float16* main_h  = attn_h + nAttn;
  _Float16* proj2_h = main_h + nMain;
  _Float16* proj3_h = proj2_h + nAttn;
  _Float16* W2p     = proj3_h + nAttn;
  _Float16* W3p     = W2p + nW;
  float*    b2p     = (float*)(W3p + nW);        // 4B-aligned (even half count)
  float*    b3p     = b2p + DIMP;                // total ws ~52.9 MB

  // ---- padding / f16 conversion of all GEMM operands
  cvt_w_kernel<<<(DIMP * DIMP + 255) / 256, 256, 0, stream>>>(W2, W2p);
  cvt_w_kernel<<<(DIMP * DIMP + 255) / 256, 256, 0, stream>>>(W3, W3p);
  cvt_bias_kernel<<<1, DIMP, 0, stream>>>(b2, b2p);
  cvt_bias_kernel<<<1, DIMP, 0, stream>>>(b3, b3p);
  {
    int rows = NB * T2, total = rows * DIMP;
    cvt_pad_kernel<<<(total + 255) / 256, 256, 0, stream>>>(input2, attn_h, rows);
  }
  {
    int rows = NB * T1, total = rows * DIMP;
    cvt_pad_kernel<<<(total + 255) / 256, 256, 0, stream>>>(input1, main_h, rows);
  }

  // ---- proj GEMMs (WMMA f16 -> f32 -> f16)
  proj_kernel<<<dim3(NB * T2 / 16, 5), 128, 0, stream>>>(attn_h, W2p, b2p, proj2_h);
  proj_kernel<<<dim3(NB * T2 / 16, 5), 128, 0, stream>>>(attn_h, W3p, b3p, proj3_h);

  // ---- fused flash attention for both projections
  flash_attn_kernel<<<dim3(T1 / 64, NB, 2), 128, 0, stream>>>(
      main_h, attn_h, proj2_h, proj3_h, out);
}
